// SelfAttention_13606456393843
// MI455X (gfx1250) — matrix operand order repaired and checked by reading, not compiled, not run on hardware
//
#include <hip/hip_runtime.h>
#include <hip/hip_bf16.h>
#include <math.h>

typedef __attribute__((ext_vector_type(16))) __bf16 v16bf;
typedef __attribute__((ext_vector_type(8)))  float  v8f;

#define BB 4
#define TT 2048
#define KD 128
#define HH 8
#define QKSCALE 0.29730177875068026f   // 128^-0.25

__device__ __forceinline__ v8f wmma_bf16(v16bf a, v16bf b, v8f c) {
    return __builtin_amdgcn_wmma_f32_16x16x32_bf16(
        /*neg_a=*/false, a, /*neg_b=*/false, b,
        /*c_mod=*/(short)0, c, /*reuse_a=*/false, /*reuse_b=*/false);
}

// A fragment (16x32 bf16): lane l holds row M = l&15; halfsel = l>>4.
// elements 0..7  -> K = k0 + 8*halfsel + e
// elements 8..15 -> K = k0 + 16 + 8*halfsel + (e-8)
__device__ __forceinline__ v16bf load_a_frag(const __bf16* __restrict__ base,
                                             int stride, int k0) {
    int lane = threadIdx.x & 31;
    int row = lane & 15;
    int hs  = lane >> 4;
    const __bf16* p = base + row * stride + k0 + hs * 8;
    v16bf f;
#pragma unroll
    for (int e = 0; e < 8; ++e) f[e] = p[e];
#pragma unroll
    for (int e = 0; e < 8; ++e) f[8 + e] = p[16 + e];
    return f;
}

// B fragment (32x16 bf16) from a [col][K] image:
// lane l holds col N = col0 + (l&15); elements 0..7 -> K = k0 + 8*(l>>4) + e,
// elements 8..15 -> K = k0 + 16 + 8*(l>>4) + (e-8)
__device__ __forceinline__ v16bf load_b_frag(const __bf16* __restrict__ base,
                                             int stride, int col0, int k0) {
    int lane = threadIdx.x & 31;
    int col = col0 + (lane & 15);
    int hs  = lane >> 4;
    const __bf16* p = base + col * stride + k0 + hs * 8;
    v16bf f;
#pragma unroll
    for (int e = 0; e < 8; ++e) f[e] = p[e];
#pragma unroll
    for (int e = 0; e < 8; ++e) f[8 + e] = p[16 + e];
    return f;
}

// ---------------------------------------------------------------------------
// Kernel 1: fused QKV projection.  X[8192,128]f32 @ W[128,1024]f32 (x3).
// Block = 128 thr (4 waves) -> 64x64 tile.  Output bf16 in [b][h][t][d],
// q and k pre-scaled by 128^-0.25.
// ---------------------------------------------------------------------------
__global__ void __launch_bounds__(128)
qkv_kernel(const float* __restrict__ x,
           const float* __restrict__ Wq, const float* __restrict__ Wk,
           const float* __restrict__ Wv,
           __bf16* __restrict__ qo, __bf16* __restrict__ ko,
           __bf16* __restrict__ vo) {
    __shared__ __align__(16) __bf16 ldsA[64][KD + 8];   // x tile, row-major
    __shared__ __align__(16) __bf16 ldsB[64][KD + 8];   // W chunk, [col][k]

    int mtile = blockIdx.x;            // 0..127  (rows of 64)
    int nblk  = blockIdx.y;            // 0..47   (16 per weight)
    int which = nblk >> 4;
    int n0    = (nblk & 15) * 64;
    const float* W = (which == 0) ? Wq : (which == 1) ? Wk : Wv;
    __bf16* dst    = (which == 0) ? qo : (which == 1) ? ko : vo;
    float oscale   = (which == 2) ? 1.0f : QKSCALE;

    int tid = threadIdx.x;

    // X tile -> LDS (f32 -> bf16), vectorized float4 reads
    const float* xsrc = x + (size_t)mtile * 64 * KD;
#pragma unroll
    for (int i = 0; i < 16; ++i) {
        int idx4 = tid + i * 128;             // 2048 float4 = 8192 floats
        int r = idx4 >> 5, c4 = idx4 & 31;    // 32 float4 per 128-col row
        float4 fv = ((const float4*)xsrc)[idx4];
        ldsA[r][c4 * 4 + 0] = (__bf16)fv.x;
        ldsA[r][c4 * 4 + 1] = (__bf16)fv.y;
        ldsA[r][c4 * 4 + 2] = (__bf16)fv.z;
        ldsA[r][c4 * 4 + 3] = (__bf16)fv.w;
    }
    // W chunk (128 k x 64 n) -> LDS transposed [n][k]
#pragma unroll
    for (int i = 0; i < 16; ++i) {
        int idx4 = tid + i * 128;             // 2048 float4
        int kk = idx4 >> 4, c4 = idx4 & 15;   // 16 float4 per 64-col row
        float4 fv = *(const float4*)&W[(size_t)kk * (KD * HH) + n0 + c4 * 4];
        ldsB[c4 * 4 + 0][kk] = (__bf16)fv.x;
        ldsB[c4 * 4 + 1][kk] = (__bf16)fv.y;
        ldsB[c4 * 4 + 2][kk] = (__bf16)fv.z;
        ldsB[c4 * 4 + 3][kk] = (__bf16)fv.w;
    }
    __syncthreads();

    int wave = tid >> 5;
    v8f acc[4];
#pragma unroll
    for (int j = 0; j < 4; ++j)
#pragma unroll
        for (int e = 0; e < 8; ++e) acc[j][e] = 0.0f;

#pragma unroll
    for (int kc = 0; kc < 4; ++kc) {
        v16bf a = load_a_frag(&ldsA[wave * 16][0], KD + 8, kc * 32);
#pragma unroll
        for (int j = 0; j < 4; ++j) {
            v16bf b = load_b_frag(&ldsB[0][0], KD + 8, j * 16, kc * 32);
            acc[j] = wmma_bf16(a, b, acc[j]);
        }
    }

    // C layout: lane holds N = lane&15, rows M = r + 8*(lane>>4)
    int lane = tid & 31, hs = lane >> 4, nc = lane & 15;
#pragma unroll
    for (int j = 0; j < 4; ++j) {
        int ng = n0 + j * 16 + nc;
        int h = ng >> 7, kd = ng & 127;
#pragma unroll
        for (int r = 0; r < 8; ++r) {
            int m = mtile * 64 + wave * 16 + r + 8 * hs;
            int bb = m >> 11, t = m & 2047;
            dst[(((size_t)(bb * HH + h)) * TT + t) * KD + kd] =
                (__bf16)(acc[j][r] * oscale);
        }
    }
}

// ---------------------------------------------------------------------------
// Kernel 2: flash attention, one (b*h, 64-query tile) per block.
// ---------------------------------------------------------------------------
__global__ void __launch_bounds__(128)
attn_kernel(const __bf16* __restrict__ q, const __bf16* __restrict__ k,
            const __bf16* __restrict__ vv, __bf16* __restrict__ o) {
    __shared__ __align__(16) __bf16 ldsK[64][KD + 8];    // key tile [t][d]
    __shared__ __align__(16) __bf16 ldsV[KD][64 + 8];    // value tile [d][t]
    __shared__ __align__(16) __bf16 ldsP[4][16][64 + 8]; // per-wave P scratch

    int bh = blockIdx.x;               // 0..31
    int qt = blockIdx.y;               // 0..31
    const __bf16* qb = q  + (size_t)bh * TT * KD;
    const __bf16* kb = k  + (size_t)bh * TT * KD;
    const __bf16* vb = vv + (size_t)bh * TT * KD;

    int tid = threadIdx.x, wave = tid >> 5, lane = tid & 31;
    int hs = lane >> 4, nc = lane & 15;

    // Q fragments for this wave's 16 rows stay in registers (strided global)
    v16bf qa[4];
    const __bf16* qrow = qb + ((size_t)qt * 64 + wave * 16) * KD;
#pragma unroll
    for (int kc = 0; kc < 4; ++kc) qa[kc] = load_a_frag(qrow, KD, kc * 32);

    v8f oacc[8];
#pragma unroll
    for (int j = 0; j < 8; ++j)
#pragma unroll
        for (int e = 0; e < 8; ++e) oacc[j][e] = 0.0f;
    float mrow[8], lrow[8];
#pragma unroll
    for (int r = 0; r < 8; ++r) { mrow[r] = -1e30f; lrow[r] = 0.0f; }

    for (int kt = 0; kt < TT / 64; ++kt) {
        const __bf16* ksrc = kb + (size_t)kt * 64 * KD;
        const __bf16* vsrc = vb + (size_t)kt * 64 * KD;
        if (kt + 1 < TT / 64) {   // gfx1250 global_prefetch_b8 for next tiles
            __builtin_prefetch(ksrc + 64 * KD + tid * 64, 0, 1);
            __builtin_prefetch(vsrc + 64 * KD + tid * 64, 0, 1);
        }
        // K tile -> LDS row-major; V tile -> LDS transposed
#pragma unroll
        for (int i = 0; i < 8; ++i) {
            int idx8 = tid + i * 128;           // 1024 x uint4 (8 bf16 each)
            int r = idx8 >> 4, c8 = idx8 & 15;  // 16 uint4 per 128-elem row
            *(uint4*)&ldsK[r][c8 * 8] = ((const uint4*)ksrc)[idx8];
            union { uint4 u; __bf16 h[8]; } tv;
            tv.u = ((const uint4*)vsrc)[idx8];
#pragma unroll
            for (int e = 0; e < 8; ++e) ldsV[c8 * 8 + e][r] = tv.h[e];
        }
        __syncthreads();

        // S = Q @ K^T  (16x64 per wave)
        v8f s[4];
#pragma unroll
        for (int j = 0; j < 4; ++j)
#pragma unroll
            for (int e = 0; e < 8; ++e) s[j][e] = 0.0f;
#pragma unroll
        for (int kc = 0; kc < 4; ++kc) {
#pragma unroll
            for (int j = 0; j < 4; ++j) {
                v16bf b = load_b_frag(&ldsK[0][0], KD + 8, j * 16, kc * 32);
                s[j] = wmma_bf16(qa[kc], b, s[j]);
            }
        }

        // online softmax (row r lives in vgpr slot r across 16-lane group)
#pragma unroll
        for (int r = 0; r < 8; ++r) {
            float rm = -1e30f;
#pragma unroll
            for (int j = 0; j < 4; ++j) rm = fmaxf(rm, s[j][r]);
            rm = fmaxf(rm, __shfl_xor(rm, 1, 32));
            rm = fmaxf(rm, __shfl_xor(rm, 2, 32));
            rm = fmaxf(rm, __shfl_xor(rm, 4, 32));
            rm = fmaxf(rm, __shfl_xor(rm, 8, 32));
            float mn = fmaxf(mrow[r], rm);
            float corr = __expf(mrow[r] - mn);
            mrow[r] = mn;
            lrow[r] *= corr;
#pragma unroll
            for (int j = 0; j < 8; ++j) oacc[j][r] *= corr;
            float ps = 0.0f;
#pragma unroll
            for (int j = 0; j < 4; ++j) {
                float p = __expf(s[j][r] - mn);
                s[j][r] = p;
                ps += p;
            }
            ps += __shfl_xor(ps, 1, 32);
            ps += __shfl_xor(ps, 2, 32);
            ps += __shfl_xor(ps, 4, 32);
            ps += __shfl_xor(ps, 8, 32);
            lrow[r] += ps;
            // spill P row to per-wave LDS scratch (A-layout source)
#pragma unroll
            for (int j = 0; j < 4; ++j)
                ldsP[wave][r + 8 * hs][j * 16 + nc] = (__bf16)s[j][r];
        }

        // O += P @ V  (P:16x64, V:64x128)
#pragma unroll
        for (int kc = 0; kc < 2; ++kc) {
            v16bf pa = load_a_frag(&ldsP[wave][0][0], 64 + 8, kc * 32);
#pragma unroll
            for (int j = 0; j < 8; ++j) {
                v16bf b = load_b_frag(&ldsV[0][0], 64 + 8, j * 16, kc * 32);
                oacc[j] = wmma_bf16(pa, b, oacc[j]);
            }
        }
        __syncthreads();
    }

    // normalize and store as [b][t][h*d] bf16 for the output projection
    int bb = bh >> 3, h = bh & 7;
#pragma unroll
    for (int j = 0; j < 8; ++j) {
        int d = j * 16 + nc;
#pragma unroll
        for (int r = 0; r < 8; ++r) {
            int t = qt * 64 + wave * 16 + r + 8 * hs;
            float val = oacc[j][r] / lrow[r];
            o[(((size_t)bb * TT + t) * (HH * KD)) + h * KD + d] = (__bf16)val;
        }
    }
}

// ---------------------------------------------------------------------------
// Kernel 3: output projection  attn[8192,1024]bf16 @ Wu[1024,128]f32 + bu
// ---------------------------------------------------------------------------
__global__ void __launch_bounds__(128)
proj_kernel(const __bf16* __restrict__ a, const float* __restrict__ Wu,
            const float* __restrict__ bu, float* __restrict__ out) {
    __shared__ __align__(16) __bf16 ldsA[64][128 + 8];
    __shared__ __align__(16) __bf16 ldsB[64][128 + 8];   // Wu chunk [n][k]

    int mtile = blockIdx.x;        // 0..127
    int n0    = blockIdx.y * 64;   // 0 or 64
    int tid = threadIdx.x, wave = tid >> 5, lane = tid & 31;
    int hs = lane >> 4, nc = lane & 15;

    v8f acc[4];
#pragma unroll
    for (int j = 0; j < 4; ++j)
#pragma unroll
        for (int e = 0; e < 8; ++e) acc[j][e] = 0.0f;

    for (int kc = 0; kc < 8; ++kc) {      // K = 1024 in chunks of 128
        const __bf16* asrc = a + ((size_t)mtile * 64) * 1024 + kc * 128;
#pragma unroll
        for (int i = 0; i < 8; ++i) {
            int idx8 = tid + i * 128;
            int r = idx8 >> 4, c8 = idx8 & 15;
            *(uint4*)&ldsA[r][c8 * 8] =
                *(const uint4*)&asrc[(size_t)r * 1024 + c8 * 8];
        }
#pragma unroll
        for (int i = 0; i < 16; ++i) {
            int idx4 = tid + i * 128;             // 2048 float4
            int kk = idx4 >> 4, c4 = idx4 & 15;   // 16 float4 per 64-col row
            float4 fv = *(const float4*)&Wu[(size_t)(kc * 128 + kk) * 128 +
                                            n0 + c4 * 4];
            ldsB[c4 * 4 + 0][kk] = (__bf16)fv.x;
            ldsB[c4 * 4 + 1][kk] = (__bf16)fv.y;
            ldsB[c4 * 4 + 2][kk] = (__bf16)fv.z;
            ldsB[c4 * 4 + 3][kk] = (__bf16)fv.w;
        }
        __syncthreads();
#pragma unroll
        for (int k2 = 0; k2 < 4; ++k2) {
            v16bf afr = load_a_frag(&ldsA[wave * 16][0], 128 + 8, k2 * 32);
#pragma unroll
            for (int j = 0; j < 4; ++j) {
                v16bf b = load_b_frag(&ldsB[0][0], 128 + 8, j * 16, k2 * 32);
                acc[j] = wmma_bf16(afr, b, acc[j]);
            }
        }
        __syncthreads();
    }

#pragma unroll
    for (int j = 0; j < 4; ++j) {
        int col = n0 + j * 16 + nc;
        float bias = bu[col];
#pragma unroll
        for (int r = 0; r < 8; ++r) {
            int row = mtile * 64 + wave * 16 + r + 8 * hs;
            out[(size_t)row * KD + col] = acc[j][r] + bias;
        }
    }
}

// ---------------------------------------------------------------------------
extern "C" void kernel_launch(void* const* d_in, const int* in_sizes, int n_in,
                              void* d_out, int out_size, void* d_ws,
                              size_t ws_size, hipStream_t stream) {
    const float* x  = (const float*)d_in[0];
    const float* Wq = (const float*)d_in[1];
    const float* Wk = (const float*)d_in[2];
    const float* Wv = (const float*)d_in[3];
    const float* Wu = (const float*)d_in[4];
    const float* bu = (const float*)d_in[5];

    size_t per = (size_t)BB * HH * TT * KD;      // 8,388,608 elems
    __bf16* qws = (__bf16*)d_ws;
    __bf16* kws = qws + per;
    __bf16* vws = kws + per;
    __bf16* aws = vws + per;                     // [8192,1024] attn output

    dim3 blk(128);
    qkv_kernel<<<dim3(128, 48), blk, 0, stream>>>(x, Wq, Wk, Wv, qws, kws, vws);
    attn_kernel<<<dim3(32, 32), blk, 0, stream>>>(qws, kws, vws, aws);
    proj_kernel<<<dim3(128, 2), blk, 0, stream>>>(aws, Wu, bu, (float*)d_out);
}
